// HAttention_28114855919647
// MI455X (gfx1250) — matrix operand order repaired and checked by reading, not compiled, not run on hardware
//
#include <hip/hip_runtime.h>
#include <hip/hip_bf16.h>
#include <stdint.h>

typedef __attribute__((ext_vector_type(16))) __bf16 v16bf;
typedef __attribute__((ext_vector_type(8)))  float  v8f;

#define EPSF 1e-7f
constexpr int B_ = 8, N_ = 1024, DIN_ = 256, DK_ = 64, H_ = 4;

// ---------------- K1: logmap0(x) -> xt (bf16) ----------------
__global__ void k_logmap(const float* __restrict__ x, __bf16* __restrict__ xtb) {
  int row  = blockIdx.x * (blockDim.x >> 5) + (threadIdx.x >> 5); // b*N+n
  int lane = threadIdx.x & 31;
  const float* xr = x + (size_t)row * DIN_;
  float v[8]; float ss = 0.f;
  #pragma unroll
  for (int k = 0; k < 8; ++k) { v[k] = xr[lane + 32*k]; ss += v[k]*v[k]; }
  #pragma unroll
  for (int m = 16; m; m >>= 1) ss += __shfl_xor(ss, m, 32);
  float nn = sqrtf(ss);
  float nc = fminf(fmaxf(nn, EPSF), 1.0f - 1e-5f);
  float scale = 0.5f * logf((1.f + nc) / (1.f - nc)) / nc;  // artanh(nc)/nc
  __bf16* xb = xtb + (size_t)row * DIN_;
  #pragma unroll
  for (int k = 0; k < 8; ++k) xb[lane + 32*k] = (__bf16)(scale * v[k]);
}

// ---------------- K2: weights -> bf16 ----------------
__global__ void k_w2bf(const float* __restrict__ Wq, const float* __restrict__ Wv,
                       __bf16* __restrict__ Wqb, __bf16* __restrict__ Wvb, int n) {
  int i = blockIdx.x * blockDim.x + threadIdx.x;
  if (i < n) { Wqb[i] = (__bf16)Wq[i]; Wvb[i] = (__bf16)Wv[i]; }
}

// ---------------- K3: h = xt @ W[h]  (WMMA bf16) -> f32 + bf16 copies ----------------
__global__ void k_gemm(const __bf16* __restrict__ xtb, const __bf16* __restrict__ Wqb,
                       const __bf16* __restrict__ Wvb, float* __restrict__ hq,
                       float* __restrict__ hv, __bf16* __restrict__ hqb,
                       __bf16* __restrict__ hvb) {
  const int tile = blockIdx.x, h = blockIdx.y, layer = blockIdx.z;
  const __bf16* Wb = layer ? Wvb : Wqb;
  float*  out  = layer ? hv  : hq;
  __bf16* outb = layer ? hvb : hqb;
  const int b = tile >> 6, n0 = (tile & 63) << 4;
  const int lane = threadIdx.x & 31, wv = threadIdx.x >> 5;
  const int e0 = wv * 16, hf = lane >> 4, r = lane & 15;
  const __bf16* arow = xtb + (size_t)(b * N_ + n0 + r) * DIN_;
  v8f acc = {};
  for (int k0 = 0; k0 < DIN_; k0 += 32) {
    union { v16bf v; uint4 q[2]; } ua, ub;
    ua.q[0] = *(const uint4*)(arow + k0 + hf * 8);
    ua.q[1] = *(const uint4*)(arow + k0 + 16 + hf * 8);
    #pragma unroll
    for (int j = 0; j < 16; ++j) {
      int kk = (j < 8) ? (hf * 8 + j) : (16 + hf * 8 + (j - 8));
      ub.v[j] = Wb[((size_t)h * DIN_ + (k0 + kk)) * DK_ + e0 + r];
    }
    acc = __builtin_amdgcn_wmma_f32_16x16x32_bf16(false, ua.v, false, ub.v,
                                                  (short)0, acc, false, false);
  }
  #pragma unroll
  for (int v = 0; v < 8; ++v) {
    int M = hf * 8 + v;
    size_t idx = (((size_t)(b * N_ + n0 + M)) * H_ + h) * DK_ + e0 + r;
    out[idx]  = acc[v];
    outb[idx] = (__bf16)acc[v];
  }
}

// ---------------- K4: es/ed dot products for both layers ----------------
__global__ void k_srcdst(const float* __restrict__ hq, const float* __restrict__ hv,
                         const float* __restrict__ aqs, const float* __restrict__ aqd,
                         const float* __restrict__ avs, const float* __restrict__ avd,
                         float* esq, float* edq, float* esv, float* edv) {
  int i = blockIdx.x * blockDim.x + threadIdx.x;   // (b*N+n)*H + h
  if (i >= B_ * N_ * H_) return;
  int h = i & 3;
  const float* q = hq + (size_t)i * DK_;
  const float* v = hv + (size_t)i * DK_;
  float s1 = 0, s2 = 0, s3 = 0, s4 = 0;
  for (int e = 0; e < DK_; ++e) {
    float qe = q[e], ve = v[e];
    s1 += qe * aqs[h * DK_ + e]; s2 += qe * aqd[h * DK_ + e];
    s3 += ve * avs[h * DK_ + e]; s4 += ve * avd[h * DK_ + e];
  }
  esq[i] = s1; edq[i] = s2; esv[i] = s3; edv[i] = s4;
}

// ---------------- K5: online softmax stats (max, sumexp) per row, 8 combos ----------------
__global__ void k_stats(const float* __restrict__ adj,
                        const float* esq, const float* edq,
                        const float* esv, const float* edv,
                        float* mxq, float* smq, float* mxv, float* smv) {
  int row  = blockIdx.x * (blockDim.x >> 5) + (threadIdx.x >> 5);  // b*N+n
  int lane = threadIdx.x & 31;
  int b = row >> 10;
  float es[8];
  #pragma unroll
  for (int h = 0; h < 4; ++h) {
    es[h]     = esq[(size_t)row * H_ + h];
    es[4 + h] = esv[(size_t)row * H_ + h];
  }
  float mx[8], sm[8];
  #pragma unroll
  for (int c = 0; c < 8; ++c) { mx[c] = -1e30f; sm[c] = 0.f; }
  const float* arow = adj + (size_t)row * N_;
  for (int m = lane; m < N_; m += 32) {
    float a = arow[m];
    if (a > 0.f) {
      size_t mi = ((size_t)(b * N_ + m)) * H_;
      #pragma unroll
      for (int c = 0; c < 8; ++c) {
        float ed = (c < 4) ? edq[mi + c] : edv[mi + (c - 4)];
        float ee = es[c] + ed;
        ee = ee > 0.f ? ee : 0.2f * ee;              // leaky_relu 0.2
        if (ee > mx[c]) { sm[c] = sm[c] * __expf(mx[c] - ee) + 1.f; mx[c] = ee; }
        else             sm[c] += __expf(ee - mx[c]);
      }
    }
  }
  #pragma unroll
  for (int off = 16; off; off >>= 1) {
    #pragma unroll
    for (int c = 0; c < 8; ++c) {
      float om = __shfl_xor(mx[c], off, 32);
      float os = __shfl_xor(sm[c], off, 32);
      float nm = fmaxf(mx[c], om);
      sm[c] = sm[c] * __expf(mx[c] - nm) + os * __expf(om - nm);
      mx[c] = nm;
    }
  }
  if (lane == 0) {
    #pragma unroll
    for (int h = 0; h < 4; ++h) {
      mxq[(size_t)row * H_ + h] = mx[h];     smq[(size_t)row * H_ + h] = sm[h];
      mxv[(size_t)row * H_ + h] = mx[4 + h]; smv[(size_t)row * H_ + h] = sm[4 + h];
    }
  }
}

// ---------------- K6: out = softmax(att) @ h  (WMMA bf16, att built in-register) --------
__global__ void k_apply(const float* __restrict__ adj,
                        const __bf16* __restrict__ hqb, const __bf16* __restrict__ hvb,
                        const float* esq, const float* edq, const float* mxq, const float* smq,
                        const float* esv, const float* edv, const float* mxv, const float* smv,
                        float* __restrict__ outq, float* __restrict__ outv) {
  const int tile = blockIdx.x, h = blockIdx.y, layer = blockIdx.z;
  const __bf16* hsrc = layer ? hvb : hqb;
  const float* es = layer ? esv : esq;
  const float* ed = layer ? edv : edq;
  const float* mx = layer ? mxv : mxq;
  const float* sm = layer ? smv : smq;
  float* out = layer ? outv : outq;
  const int b = tile >> 6, n0 = (tile & 63) << 4;
  const int lane = threadIdx.x & 31, wv = threadIdx.x >> 5;
  const int e0 = wv * 16, hf = lane >> 4, r = lane & 15;
  const int n = n0 + r;
  const float es_n = es[(size_t)(b * N_ + n) * H_ + h];
  const float mx_n = mx[(size_t)(b * N_ + n) * H_ + h];
  const float sv   = sm[(size_t)(b * N_ + n) * H_ + h];
  const float inv_n = sv > 0.f ? 1.f / sv : 0.f;
  const float* arow = adj + (size_t)(b * N_ + n) * N_;
  v8f acc = {};
  for (int mt = 0; mt < N_ / 32; ++mt) {
    union { v16bf v; __bf16 e[16]; uint4 q[2]; } ua, ub;
    #pragma unroll
    for (int j = 0; j < 16; ++j) {
      int k = (j < 8) ? (hf * 8 + j) : (16 + hf * 8 + (j - 8));
      int m = mt * 32 + k;
      float a = arow[m];
      float val = 0.f;
      if (a > 0.f) {
        float ee = es_n + ed[(size_t)(b * N_ + m) * H_ + h];
        ee = ee > 0.f ? ee : 0.2f * ee;
        val = __expf(ee - mx_n) * inv_n;
      }
      ua.e[j] = (__bf16)val;
    }
    #pragma unroll
    for (int j = 0; j < 16; ++j) {
      int mb = mt * 32 + ((j < 8) ? (hf * 8 + j) : (16 + hf * 8 + (j - 8)));
      ub.e[j] = hsrc[((size_t)(b * N_ + mb) * H_ + h) * DK_ + e0 + r];
    }
    acc = __builtin_amdgcn_wmma_f32_16x16x32_bf16(false, ua.v, false, ub.v,
                                                  (short)0, acc, false, false);
  }
  #pragma unroll
  for (int v = 0; v < 8; ++v) {
    int M = hf * 8 + v;
    out[(((size_t)(b * N_ + n0 + M)) * H_ + h) * DK_ + e0 + r] = acc[v];
  }
}

// ---------------- K7a: query = expmap0(outq) per head (bf16) + ||q||^2 ----------------
__global__ void k_expq(const float* __restrict__ outq, __bf16* __restrict__ qb,
                       float* __restrict__ sqq) {
  int i    = blockIdx.x * (blockDim.x >> 5) + (threadIdx.x >> 5);  // (b*N+n)*H+h
  int lane = threadIdx.x & 31;
  const float* v = outq + (size_t)i * DK_;
  float a = v[lane], b2 = v[lane + 32];
  float ss = a * a + b2 * b2;
  #pragma unroll
  for (int m = 16; m; m >>= 1) ss += __shfl_xor(ss, m, 32);
  float nn = fmaxf(sqrtf(ss), EPSF);
  float scale = tanhf(nn) / nn;
  __bf16* qo = qb + (size_t)i * DK_;
  qo[lane]      = (__bf16)(scale * a);
  qo[lane + 32] = (__bf16)(scale * b2);
  if (lane == 0) sqq[i] = scale * scale * ss;
}

// ---------------- K7b: value = expmap0(concat heads of outv) -> d_out ----------------
__global__ void k_expv(const float* __restrict__ outv, float* __restrict__ val) {
  int row  = blockIdx.x * (blockDim.x >> 5) + (threadIdx.x >> 5);  // b*N+n
  int lane = threadIdx.x & 31;
  const float* v = outv + (size_t)row * 256;
  float x[8]; float ss = 0.f;
  #pragma unroll
  for (int k = 0; k < 8; ++k) { x[k] = v[lane + 32*k]; ss += x[k]*x[k]; }
  #pragma unroll
  for (int m = 16; m; m >>= 1) ss += __shfl_xor(ss, m, 32);
  float nn = fmaxf(sqrtf(ss), EPSF);
  float scale = tanhf(nn) / nn;
  float* o = val + (size_t)row * 256;
  #pragma unroll
  for (int k = 0; k < 8; ++k) o[lane + 32*k] = scale * x[k];
}

// ---------------- K7c: transpose query -> qT[b,h,k,j] (bf16, K-major for B-frags) ----
__global__ void k_qt(const __bf16* __restrict__ qb, __bf16* __restrict__ qT) {
  int i = blockIdx.x * blockDim.x + threadIdx.x;
  if (i >= B_ * H_ * DK_ * N_) return;
  int j = i & (N_ - 1);
  int k = (i >> 10) & (DK_ - 1);
  int h = (i >> 16) & (H_ - 1);
  int b = i >> 18;
  qT[i] = qb[(((size_t)(b * N_ + j)) * H_ + h) * DK_ + k];
}

// ---------------- K8: Q@Q^T (WMMA) -> arccosh dist -> tanh -> head-mean -> att ------
__global__ void k_dist(const __bf16* __restrict__ qb, const __bf16* __restrict__ qT,
                       const float* __restrict__ sqq,
                       const float* __restrict__ rr, const float* __restrict__ tt,
                       float* __restrict__ att) {
  const int b  = blockIdx.z;
  const int i0 = blockIdx.x * 16;
  const int jt = blockIdx.y * (blockDim.x >> 5) + (threadIdx.x >> 5);
  const int j0 = jt * 16;
  const int lane = threadIdx.x & 31, hf = lane >> 4, r = lane & 15;
  v8f attacc = {};
  for (int h = 0; h < H_; ++h) {
    v8f dot = {};
    const __bf16* arow = qb + (((size_t)(b * N_ + i0 + r)) * H_ + h) * DK_;
    #pragma unroll
    for (int kt = 0; kt < 2; ++kt) {
      int k0 = kt * 32;
      union { v16bf v; uint4 q[2]; } ua, ub;
      ua.q[0] = *(const uint4*)(arow + k0 + hf * 8);
      ua.q[1] = *(const uint4*)(arow + k0 + 16 + hf * 8);
      const __bf16* brow =
          qb + (((size_t)(b * N_ + j0 + r)) * H_ + h) * DK_;
      ub.q[0] = *(const uint4*)(brow + k0 + hf * 8);
      ub.q[1] = *(const uint4*)(brow + k0 + 16 + hf * 8);
      dot = __builtin_amdgcn_wmma_f32_16x16x32_bf16(false, ua.v, false, ub.v,
                                                    (short)0, dot, false, false);
    }
    float rh = rr[h], th = tt[h];
    float sqj = sqq[((size_t)(b * N_ + j0 + r)) * H_ + h];
    #pragma unroll
    for (int v = 0; v < 8; ++v) {
      int i = i0 + hf * 8 + v;
      float sqi = sqq[((size_t)(b * N_ + i)) * H_ + h];
      float d2  = fmaxf(sqi + sqj - 2.f * dot[v], 0.f);
      float den = fmaxf((1.f - sqi) * (1.f - sqj), EPSF);
      float z   = fmaxf(1.f + 2.f * d2 / den, 1.f + EPSF);
      float dd  = logf(z + sqrtf(z * z - 1.f));        // arccosh
      attacc[v] += tanhf((rh - dd) * th);
    }
  }
  #pragma unroll
  for (int v = 0; v < 8; ++v) {
    int i = i0 + hf * 8 + v;
    att[((size_t)b * N_ + i) * N_ + j0 + r] = attacc[v] * 0.25f;
  }
}

extern "C" void kernel_launch(void* const* d_in, const int* in_sizes, int n_in,
                              void* d_out, int out_size, void* d_ws, size_t ws_size,
                              hipStream_t stream) {
  const float* x   = (const float*)d_in[0];
  const float* adj = (const float*)d_in[1];
  // d_in[2] flags: unused by reference
  const float* Wq  = (const float*)d_in[3];
  const float* aqs = (const float*)d_in[4];
  const float* aqd = (const float*)d_in[5];
  const float* Wv  = (const float*)d_in[6];
  const float* avs = (const float*)d_in[7];
  const float* avd = (const float*)d_in[8];
  const float* rr  = (const float*)d_in[9];
  const float* tt  = (const float*)d_in[10];

  float* value = (float*)d_out;                         // [B,N,256]
  float* att   = (float*)d_out + (size_t)B_ * N_ * 256; // [B,N,N]

  char* p = (char*)d_ws;
  auto carve = [&](size_t bytes) -> char* {
    char* q = p; p += (bytes + 255) & ~(size_t)255; return q;
  };
  const size_t BN = (size_t)B_ * N_;
  __bf16* xtb = (__bf16*)carve(2 * BN * DIN_);
  __bf16* Wqb = (__bf16*)carve(2 * (size_t)H_ * DIN_ * DK_);
  __bf16* Wvb = (__bf16*)carve(2 * (size_t)H_ * DIN_ * DK_);
  float*  hq  = (float*) carve(4 * BN * H_ * DK_);
  float*  hv  = (float*) carve(4 * BN * H_ * DK_);
  __bf16* hqb = (__bf16*)carve(2 * BN * H_ * DK_);
  __bf16* hvb = (__bf16*)carve(2 * BN * H_ * DK_);
  float* esq  = (float*)carve(4 * BN * H_);
  float* edq  = (float*)carve(4 * BN * H_);
  float* esv  = (float*)carve(4 * BN * H_);
  float* edv  = (float*)carve(4 * BN * H_);
  float* mxq  = (float*)carve(4 * BN * H_);
  float* smq  = (float*)carve(4 * BN * H_);
  float* mxv  = (float*)carve(4 * BN * H_);
  float* smv  = (float*)carve(4 * BN * H_);
  float* outq = (float*)carve(4 * BN * H_ * DK_);
  float* outv = (float*)carve(4 * BN * H_ * DK_);
  __bf16* qb  = (__bf16*)carve(2 * BN * H_ * DK_);
  __bf16* qT  = (__bf16*)carve(2 * BN * H_ * DK_);
  float*  sqq = (float*) carve(4 * BN * H_);

  k_logmap<<<B_ * N_ / 8, 256, 0, stream>>>(x, xtb);
  k_w2bf<<<(H_ * DIN_ * DK_ + 255) / 256, 256, 0, stream>>>(Wq, Wv, Wqb, Wvb,
                                                            H_ * DIN_ * DK_);
  k_gemm<<<dim3(B_ * N_ / 16, H_, 2), 128, 0, stream>>>(xtb, Wqb, Wvb,
                                                        hq, hv, hqb, hvb);
  k_srcdst<<<(B_ * N_ * H_ + 255) / 256, 256, 0, stream>>>(hq, hv, aqs, aqd, avs, avd,
                                                           esq, edq, esv, edv);
  k_stats<<<B_ * N_ / 8, 256, 0, stream>>>(adj, esq, edq, esv, edv,
                                           mxq, smq, mxv, smv);
  k_apply<<<dim3(B_ * N_ / 16, H_, 2), 128, 0, stream>>>(adj, hqb, hvb,
                                                         esq, edq, mxq, smq,
                                                         esv, edv, mxv, smv,
                                                         outq, outv);
  k_expq<<<B_ * N_ * H_ / 8, 256, 0, stream>>>(outq, qb, sqq);
  k_expv<<<B_ * N_ / 8, 256, 0, stream>>>(outv, value);
  k_qt<<<(B_ * H_ * DK_ * N_ + 255) / 256, 256, 0, stream>>>(qb, qT);
  k_dist<<<dim3(N_ / 16, N_ / 16 / 8, B_), 256, 0, stream>>>(qb, qT, sqq, rr, tt, att);
}
